// FullyParallelQwenAttention_58497454571736
// MI455X (gfx1250) — compile-verified, matrix operand order checked by reading, not run on hardware
//
#include <hip/hip_runtime.h>

// ---------------------------------------------------------------------------
// CDNA5 (gfx1250) bf16 WMMA attention block.
// Matmuls -> v_wmma_f32_16x16x32_bf16 (wave32 layouts per CDNA5 ISA 7.12.2).
// Tile staging -> global_load_async_to_lds_b128 (ASYNCcnt, ISA 15.18.3),
// software-pipelined against WMMA compute in the GEMM.
// ---------------------------------------------------------------------------

typedef __bf16 bf16;
typedef __attribute__((ext_vector_type(16))) __bf16 v16bf;
typedef __attribute__((ext_vector_type(8)))  float  v8f;

#define N_HEADS 16
#define N_KV    8
#define HD      128
#define SEQ     2048
#define BATCH   2
#define HID     2048

__device__ __forceinline__ float bf2f(bf16 x) {
    unsigned short s = __builtin_bit_cast(unsigned short, x);
    unsigned u = ((unsigned)s) << 16;
    return __builtin_bit_cast(float, u);
}
__device__ __forceinline__ bf16 f2bf(float f) {
    unsigned u = __builtin_bit_cast(unsigned, f);
    unsigned r = (u + 0x7FFFu + ((u >> 16) & 1u)) >> 16;   // round-nearest-even
    unsigned short s = (unsigned short)r;
    return __builtin_bit_cast(bf16, s);
}

// ---------------------------------------------------------------------------
// Async global->LDS copies (ASYNCcnt-tracked).  LDS operand = low 32 bits of
// the generic shared pointer (ISA 10.2 truncation); inst_offset applies to
// both global and LDS addresses (ISA 10.7).
// ---------------------------------------------------------------------------
__device__ __forceinline__ void async_copy32B(const bf16* g, const bf16* l) {
    unsigned lo = (unsigned)(unsigned long long)l;
    asm volatile(
        "global_load_async_to_lds_b128 %0, %1, off\n\t"
        "global_load_async_to_lds_b128 %0, %1, off offset:16"
        :: "v"(lo), "v"(g) : "memory");
}
__device__ __forceinline__ void async_copy64B(const bf16* g, const bf16* l) {
    unsigned lo = (unsigned)(unsigned long long)l;
    asm volatile(
        "global_load_async_to_lds_b128 %0, %1, off\n\t"
        "global_load_async_to_lds_b128 %0, %1, off offset:16\n\t"
        "global_load_async_to_lds_b128 %0, %1, off offset:32\n\t"
        "global_load_async_to_lds_b128 %0, %1, off offset:48"
        :: "v"(lo), "v"(g) : "memory");
}
__device__ __forceinline__ void wait_async() {
    asm volatile("s_wait_asynccnt 0x0" ::: "memory");
}

// Load one 16x32 bf16 fragment (A-operand layout, CDNA5 ISA 7.12.2):
// lane l holds row (l&15); dword v holds K = 2*(v%4) + 8*((l>>4) + 2*(v/4)).
// Same pattern serves the B operand (row index == N column, K contiguous).
__device__ __forceinline__ v16bf load_frag(const bf16* base, int ldw) {
    const int l = threadIdx.x & 31;
    const int r = l & 15;
    const int g = l >> 4;
    union { v16bf v; unsigned u[8]; } f;
#pragma unroll
    for (int i = 0; i < 8; ++i) {
        int kk = ((i & 3) << 1) + ((g + ((i >> 2) << 1)) << 3);
        f.u[i] = *(const unsigned*)(base + (size_t)r * ldw + kk);
    }
    return f.v;
}

// ---------------------------------------------------------------------------
// one-time conversions / re-layouts
// ---------------------------------------------------------------------------
__global__ void cvt_kernel(const float* __restrict__ src, bf16* __restrict__ dst, int n) {
    int i = blockIdx.x * blockDim.x + threadIdx.x;
    if (i < n) dst[i] = f2bf(src[i]);
}

// src[K][N] f32  ->  dst[N][K] bf16  (weights pre-transposed once)
__global__ void cvtT_kernel(const float* __restrict__ src, bf16* __restrict__ dst,
                            int K, int N) {
    size_t idx = (size_t)blockIdx.x * blockDim.x + threadIdx.x;
    int k = (int)(idx % K);
    int n = (int)(idx / K);
    if (n < N) dst[(size_t)n * K + k] = f2bf(src[(size_t)k * N + n]);
}

// v[(b*S+s)*8+kv][d]  ->  vT[((b*8+kv)*128+d)*S + s]   (V pre-transposed once)
__global__ void vT_kernel(const bf16* __restrict__ v, bf16* __restrict__ vT) {
    size_t idx = (size_t)blockIdx.x * blockDim.x + threadIdx.x;   // over B*8*128*S
    int s  = (int)(idx % SEQ);
    int d  = (int)((idx / SEQ) % HD);
    int kv = (int)((idx / ((size_t)SEQ * HD)) % N_KV);
    int b  = (int)(idx / ((size_t)SEQ * HD * N_KV));
    if (b < BATCH)
        vT[idx] = v[(((size_t)(b * SEQ + s)) * N_KV + kv) * HD + d];
}

// ---------------------------------------------------------------------------
// bf16 GEMM: C[M,N] = A[M,K] * BT[N,K]^T (+bias). Block tile 128x128, 8 waves
// (2x4), each wave 64x32 = 4x2 WMMA tiles, K step 32.  Double-buffered LDS:
// iteration i issues the async DMA of tile i+1, then computes on tile i, so
// the ASYNC copy engine runs under the WMMA pipe.
// ---------------------------------------------------------------------------
template <bool OUT_F32>
__global__ __launch_bounds__(256) void gemm_bf16_kernel(
    const bf16* __restrict__ A, const bf16* __restrict__ BT,
    const bf16* __restrict__ bias, void* __restrict__ Cout,
    int M, int N, int K)
{
    __shared__ __align__(16) bf16 As[2][128 * 40];   // stride 40 -> conflict-free
    __shared__ __align__(16) bf16 Bs[2][128 * 40];   // B tile, [n][k]

    const int t    = threadIdx.x;
    const int wid  = t >> 5;
    const int lane = t & 31;
    const int r16  = lane & 15;
    const int g    = lane >> 4;
    const int m0   = blockIdx.y * 128;
    const int n0   = blockIdx.x * 128;
    const int wm   = (wid & 1) * 64;
    const int wn   = (wid >> 1) * 32;

    v8f acc[4][2];
#pragma unroll
    for (int i = 0; i < 4; ++i)
#pragma unroll
        for (int j = 0; j < 2; ++j) acc[i][j] = {};

    const int row16 = t >> 1;            // 0..127
    const int seg16 = (t & 1) * 16;      // 0 or 16 (bf16 elems)
    const bf16* ga = A  + (size_t)(m0 + row16) * K + seg16;
    const bf16* gb = BT + (size_t)(n0 + row16) * K + seg16;
    const int lofs = row16 * 40 + seg16;

    const int niter = K >> 5;

    // prologue: stage tile 0 into buffer 0
    async_copy32B(ga, As[0] + lofs);
    async_copy32B(gb, Bs[0] + lofs);
    wait_async();
    __syncthreads();

    for (int i = 0; i < niter; ++i) {
        const bf16* as = As[i & 1];
        const bf16* bs = Bs[i & 1];
        if (i + 1 < niter) {               // DMA next tile under this compute
            async_copy32B(ga + (i + 1) * 32, As[(i + 1) & 1] + lofs);
            async_copy32B(gb + (i + 1) * 32, Bs[(i + 1) & 1] + lofs);
        }

        v16bf af[4], bfg[2];
#pragma unroll
        for (int mt = 0; mt < 4; ++mt) af[mt]  = load_frag(as + (wm + mt * 16) * 40, 40);
#pragma unroll
        for (int nt = 0; nt < 2; ++nt) bfg[nt] = load_frag(bs + (wn + nt * 16) * 40, 40);
#pragma unroll
        for (int mt = 0; mt < 4; ++mt)
#pragma unroll
            for (int nt = 0; nt < 2; ++nt)
                acc[mt][nt] = __builtin_amdgcn_wmma_f32_16x16x32_bf16(
                    false, af[mt], false, bfg[nt], (short)0, acc[mt][nt], false, false);

        wait_async();      // next tile landed
        __syncthreads();   // all waves done reading this tile; next visible
    }

    // epilogue: C layout -> lane l, vgpr j = (row j+8*(l>>4), col l&15)
#pragma unroll
    for (int mt = 0; mt < 4; ++mt) {
#pragma unroll
        for (int nt = 0; nt < 2; ++nt) {
            int col = n0 + wn + nt * 16 + r16;
            float bv = bias ? bf2f(bias[col]) : 0.0f;
#pragma unroll
            for (int j = 0; j < 8; ++j) {
                int row = m0 + wm + mt * 16 + j + 8 * g;
                float v = acc[mt][nt][j] + bv;
                if (OUT_F32) ((float*)Cout)[(size_t)row * N + col] = v;
                else         ((bf16*)Cout)[(size_t)row * N + col] = f2bf(v);
            }
        }
    }
}

// ---------------------------------------------------------------------------
// RoPE on q (B,S,16,128) and k (B,S,8,128), pair (i, i+64), in-place bf16.
// ---------------------------------------------------------------------------
__global__ void rope_kernel(bf16* __restrict__ Qb, bf16* __restrict__ Kb,
                            const int* __restrict__ pos)
{
    size_t idx = (size_t)blockIdx.x * blockDim.x + threadIdx.x;
    int i  = (int)(idx & 63);
    int hh = (int)((idx >> 6) % 24);
    int s  = (int)(((idx >> 6) / 24) % SEQ);
    int b  = (int)(idx / (64ull * 24 * SEQ));
    if (b >= BATCH) return;

    float p   = (float)pos[b * SEQ + s];
    float inv = __expf(-(float)i * (13.815510557964274f / 64.0f)); // theta=1e6
    float ang = p * inv;
    float sn, cs;
    __sincosf(ang, &sn, &cs);

    bf16* base;
    if (hh < N_HEADS) base = Qb + (((size_t)(b * SEQ + s)) * N_HEADS + hh) * HD;
    else              base = Kb + (((size_t)(b * SEQ + s)) * N_KV + (hh - N_HEADS)) * HD;
    float x0 = bf2f(base[i]), x1 = bf2f(base[i + 64]);
    base[i]      = f2bf(x0 * cs - x1 * sn);
    base[i + 64] = f2bf(x0 * sn + x1 * cs);
}

// ---------------------------------------------------------------------------
// Flash-style causal attention. Block = 128 query rows of one (b,h); 8 waves,
// each a 16-row tile; Q fragments in registers. 64-key tiles; K [key][d] and
// pre-transposed V [d][key] staged via async b128 copies; online softmax;
// P goes through per-wave LDS for C-layout -> A-fragment conversion.
// Score loop is kc-outer / nt-inner: 4 independent WMMA chains so fragment
// ds_loads pipeline under the matrix pipe.
// ---------------------------------------------------------------------------
__global__ __launch_bounds__(256) void attn_kernel(
    const bf16* __restrict__ Q, const bf16* __restrict__ Kb,
    const bf16* __restrict__ VT, bf16* __restrict__ O)
{
    __shared__ __align__(16) bf16 Ks[64 * 136];   // [key][d], stride 136
    __shared__ __align__(16) bf16 Vt[128 * 72];   // [d][key], stride 72
    __shared__ __align__(16) bf16 Pb[8 * 16 * 72];

    const int t    = threadIdx.x;
    const int wid  = t >> 5;
    const int lane = t & 31;
    const int r16  = lane & 15;
    const int g    = lane >> 4;
    const int qt   = blockIdx.x;          // query tile of 128
    const int h    = blockIdx.y;
    const int b    = blockIdx.z;
    const int kvh  = h >> 1;              // GQA: 2 q-heads per kv-head
    const int qb   = qt * 128;
    const int q0   = qb + wid * 16;

    // Q fragments for this wave's 16 rows, 4 K-chunks of 32 (hd=128)
    v16bf qf[4];
    const bf16* qbase = Q + (((size_t)(b * SEQ + q0)) * N_HEADS + h) * HD;
#pragma unroll
    for (int kc = 0; kc < 4; ++kc)
        qf[kc] = load_frag(qbase + kc * 32, N_HEADS * HD);

    v8f acc[8];
#pragma unroll
    for (int i = 0; i < 8; ++i) acc[i] = {};
    float m_i[8], l_i[8];
#pragma unroll
    for (int j = 0; j < 8; ++j) { m_i[j] = -1e30f; l_i[j] = 0.0f; }

    const float scale = 0.08838834764831845f;  // 1/sqrt(128)
    const int nkt = (qb + 128) >> 6;           // causal: only needed key tiles

    const int krow = t >> 2, kseg = (t & 3) * 32;   // K tile mapping (64B/thr)
    const int vrow = t >> 1, vseg = (t & 1) * 32;   // V tile mapping (64B/thr)
    const bf16* vtb = VT + ((size_t)(b * N_KV + kvh) * HD + vrow) * SEQ + vseg;

    for (int kt = 0; kt < nkt; ++kt) {
        const int key0 = kt * 64;
        __syncthreads();   // previous iteration's LDS reads complete
        async_copy64B(Kb + (((size_t)(b * SEQ + key0 + krow)) * N_KV + kvh) * HD + kseg,
                      Ks + krow * 136 + kseg);
        async_copy64B(vtb + key0,
                      Vt + vrow * 72 + vseg);
        wait_async();
        __syncthreads();

        // scores: 16x64; kc outer so the 4 st[] chains are independent
        v8f st[4];
#pragma unroll
        for (int nt = 0; nt < 4; ++nt) st[nt] = {};
#pragma unroll
        for (int kc = 0; kc < 4; ++kc)
#pragma unroll
            for (int nt = 0; nt < 4; ++nt) {
                v16bf kf = load_frag(Ks + (nt * 16) * 136 + kc * 32, 136);
                st[nt] = __builtin_amdgcn_wmma_f32_16x16x32_bf16(
                             false, qf[kc], false, kf, (short)0, st[nt], false, false);
            }

        // scale + causal mask + row max (rows live in 16-lane halves)
        float mnew[8];
#pragma unroll
        for (int j = 0; j < 8; ++j) {
            int qrow = q0 + j + 8 * g;
            float mx = -1e30f;
#pragma unroll
            for (int nt = 0; nt < 4; ++nt) {
                int key = key0 + nt * 16 + r16;
                float s = st[nt][j] * scale;
                if (key > qrow) s = -1e9f;
                st[nt][j] = s;
                mx = fmaxf(mx, s);
            }
            mx = fmaxf(mx, __shfl_xor(mx, 1, 32));
            mx = fmaxf(mx, __shfl_xor(mx, 2, 32));
            mx = fmaxf(mx, __shfl_xor(mx, 4, 32));
            mx = fmaxf(mx, __shfl_xor(mx, 8, 32));
            mnew[j] = fmaxf(m_i[j], mx);
        }

        // online softmax: rescale, exponentiate, row sums
#pragma unroll
        for (int j = 0; j < 8; ++j) {
            float alpha = __expf(m_i[j] - mnew[j]);
            l_i[j] *= alpha;
#pragma unroll
            for (int dt = 0; dt < 8; ++dt) acc[dt][j] *= alpha;
            float rs = 0.0f;
#pragma unroll
            for (int nt = 0; nt < 4; ++nt) {
                float p = __expf(st[nt][j] - mnew[j]);
                st[nt][j] = p;
                rs += p;
            }
            rs += __shfl_xor(rs, 1, 32);
            rs += __shfl_xor(rs, 2, 32);
            rs += __shfl_xor(rs, 4, 32);
            rs += __shfl_xor(rs, 8, 32);
            l_i[j] += rs;
            m_i[j] = mnew[j];
        }

        // P (C layout) -> LDS -> A-fragment layout
        bf16* pw = Pb + wid * 16 * 72;
#pragma unroll
        for (int nt = 0; nt < 4; ++nt)
#pragma unroll
            for (int j = 0; j < 8; ++j)
                pw[(j + 8 * g) * 72 + nt * 16 + r16] = f2bf(st[nt][j]);
        __syncthreads();

        // acc += P(16x64) x V(64x128): kc outer, 8 independent acc chains
#pragma unroll
        for (int kc = 0; kc < 2; ++kc) {
            v16bf pf = load_frag(pw + kc * 32, 72);
#pragma unroll
            for (int dt = 0; dt < 8; ++dt) {
                v16bf vf = load_frag(Vt + (dt * 16) * 72 + kc * 32, 72);
                acc[dt] = __builtin_amdgcn_wmma_f32_16x16x32_bf16(
                              false, pf, false, vf, (short)0, acc[dt], false, false);
            }
        }
    }

    // epilogue: normalize and store as (B,S,16,128) bf16
#pragma unroll
    for (int dt = 0; dt < 8; ++dt) {
#pragma unroll
        for (int j = 0; j < 8; ++j) {
            int qrow = q0 + j + 8 * g;
            int d = dt * 16 + r16;
            float v = acc[dt][j] / l_i[j];
            O[(((size_t)(b * SEQ + qrow)) * N_HEADS + h) * HD + d] = f2bf(v);
        }
    }
}

// ---------------------------------------------------------------------------
// host-side orchestration
// ---------------------------------------------------------------------------
extern "C" void kernel_launch(void* const* d_in, const int* in_sizes, int n_in,
                              void* d_out, int out_size, void* d_ws, size_t ws_size,
                              hipStream_t stream)
{
    (void)in_sizes; (void)n_in; (void)out_size; (void)ws_size;

    const float* x   = (const float*)d_in[0];
    // d_in[1] = attention_mask: causal, applied analytically in attn_kernel
    const int*   pos = (const int*)d_in[2];
    const float* wq  = (const float*)d_in[3];
    const float* bq  = (const float*)d_in[4];
    const float* wk  = (const float*)d_in[5];
    const float* bk  = (const float*)d_in[6];
    const float* wv  = (const float*)d_in[7];
    const float* bv  = (const float*)d_in[8];
    const float* wo  = (const float*)d_in[9];

    const size_t M = (size_t)BATCH * SEQ;   // 4096
    const int KV = N_KV * HD;               // 1024
    bf16* ws = (bf16*)d_ws;
    size_t o = 0;
    bf16* xb   = ws + o; o += M * HID;
    bf16* wqT  = ws + o; o += (size_t)HID * HID;     // [N][K]
    bf16* wkT  = ws + o; o += (size_t)KV * HID;
    bf16* wvT  = ws + o; o += (size_t)KV * HID;
    bf16* woT  = ws + o; o += (size_t)HID * HID;
    bf16* bqb  = ws + o; o += HID;
    bf16* bkb  = ws + o; o += KV;
    bf16* bvb  = ws + o; o += KV;
    bf16* qbuf = ws + o; o += M * HID;
    bf16* kbuf = ws + o; o += M * KV;
    bf16* vbuf = ws + o; o += M * KV;
    bf16* vTb  = ws + o; o += M * KV;                // [b][kv][d][s]
    bf16* abuf = ws + o; o += M * HID;

    cvt_kernel<<<(unsigned)((M * HID + 255) / 256), 256, 0, stream>>>(x, xb, (int)(M * HID));
    cvt_kernel<<<(HID + 255) / 256, 256, 0, stream>>>(bq, bqb, HID);
    cvt_kernel<<<(KV + 255) / 256, 256, 0, stream>>>(bk, bkb, KV);
    cvt_kernel<<<(KV + 255) / 256, 256, 0, stream>>>(bv, bvb, KV);

    cvtT_kernel<<<(unsigned)(((size_t)HID * HID + 255) / 256), 256, 0, stream>>>(wq, wqT, HID, HID);
    cvtT_kernel<<<(unsigned)(((size_t)HID * KV + 255) / 256), 256, 0, stream>>>(wk, wkT, HID, KV);
    cvtT_kernel<<<(unsigned)(((size_t)HID * KV + 255) / 256), 256, 0, stream>>>(wv, wvT, HID, KV);
    cvtT_kernel<<<(unsigned)(((size_t)HID * HID + 255) / 256), 256, 0, stream>>>(wo, woT, HID, HID);

    dim3 blk(256);
    gemm_bf16_kernel<false><<<dim3(HID / 128, (int)(M / 128)), blk, 0, stream>>>(
        xb, wqT, bqb, qbuf, (int)M, HID, HID);
    gemm_bf16_kernel<false><<<dim3(KV / 128, (int)(M / 128)), blk, 0, stream>>>(
        xb, wkT, bkb, kbuf, (int)M, KV, HID);
    gemm_bf16_kernel<false><<<dim3(KV / 128, (int)(M / 128)), blk, 0, stream>>>(
        xb, wvT, bvb, vbuf, (int)M, KV, HID);

    {   // RoPE on q and k
        size_t total = (size_t)BATCH * SEQ * (N_HEADS + N_KV) * (HD / 2);
        rope_kernel<<<(unsigned)((total + 255) / 256), 256, 0, stream>>>(qbuf, kbuf, pos);
    }

    // pre-transpose V for the PV matmul B-operand
    vT_kernel<<<(unsigned)((M * KV + 255) / 256), 256, 0, stream>>>(vbuf, vTb);

    attn_kernel<<<dim3(SEQ / 128, N_HEADS, BATCH), blk, 0, stream>>>(qbuf, kbuf, vTb, abuf);

    gemm_bf16_kernel<true><<<dim3(HID / 128, (int)(M / 128)), blk, 0, stream>>>(
        abuf, woT, nullptr, d_out, (int)M, HID, HID);
}
